// MultiHeadedAttention_83906481095437
// MI455X (gfx1250) — matrix operand order repaired and checked by reading, not compiled, not run on hardware
//
#include <hip/hip_runtime.h>
#include <hip/hip_bf16.h>

// MI455X (gfx1250) local-attention pipeline:
//   K1: QKV projection GEMMs via v_wmma_f32_16x16x32_f16 (f32->f16 in-register,
//       f32 accumulate, +bias, store Q/K/V f16 in head-major layout in d_ws)
//   K2: per (head, 16-query tile):
//         - 16x272 score tile via WMMA (17 col tiles x 2 K-steps)
//         - softmax with wave32 half-wave shfl_xor reductions
//         - probs f32 -> d_out, probs f16 -> LDS (A-fragment layout)
//         - V tile transposed into LDS (f16, zero-padded to K=288)
//         - P@V via 36 more WMMAs (9 K-steps x 4 N-tiles), f32 out store.

typedef __attribute__((ext_vector_type(16))) _Float16 v16h;
typedef __attribute__((ext_vector_type(8)))  _Float16 v8h;
typedef __attribute__((ext_vector_type(2)))  _Float16 v2h;
typedef __attribute__((ext_vector_type(8)))  float    v8f;

#define SEQ     4096
#define BATCH   4
#define HIDDEN  1024
#define NH      16
#define HD      64
#define NWIN    257
#define HALFW   128
#define NBH     (BATCH*NH)      // 64
#define MIN_SCORE -1e9f
#define VSTR    296             // LDS row stride (f16): mult of 8, 592B = 37*16

__device__ __forceinline__ v8f wmma_f16(v16h a, v16h b, v8f c) {
  // (neg_a, A, neg_b, B, c_mod, C, reuse_a, reuse_b)
  return __builtin_amdgcn_wmma_f32_16x16x32_f16(false, a, false, b, (short)0, c,
                                                false, false);
}

// ---------------------------------------------------------------------------
// Kernel 1: C[m, col] = sum_k X[m,k] * W[col,k] + bias[col]  (X = [16384,1024])
// Each wave: one 16-row M tile x one 64-col group (= one head) of one matrix.
// 3 matrices * 1024 mtiles * 16 headgroups = 49152 waves, 8 waves/block.
// ---------------------------------------------------------------------------
__global__ __launch_bounds__(256) void qkv_proj_wmma(
    const float* __restrict__ Xq, const float* __restrict__ Xk,
    const float* __restrict__ Xv,
    const float* __restrict__ Wq, const float* __restrict__ bq,
    const float* __restrict__ Wk, const float* __restrict__ bk,
    const float* __restrict__ Wv, const float* __restrict__ bv,
    _Float16* __restrict__ qh, _Float16* __restrict__ kh,
    _Float16* __restrict__ vh)
{
  const int lane = threadIdx.x & 31;
  const int l16  = lane & 15;
  const int hi   = lane >> 4;                       // 0 or 1 (half-wave)
  const int wid  = blockIdx.x * 8 + (threadIdx.x >> 5);

  const int mat   = wid >> 14;                      // 16384 wave-jobs / matrix
  const int rem   = wid & 16383;
  const int mtile = rem >> 4;
  const int ngrp  = rem & 15;                       // head index

  const float* X;  const float* Wm; const float* bias; _Float16* dst;
  if (mat == 0)      { X = Xq; Wm = Wq; bias = bq; dst = qh; }
  else if (mat == 1) { X = Xk; Wm = Wk; bias = bk; dst = kh; }
  else               { X = Xv; Wm = Wv; bias = bv; dst = vh; }

  const int abase = hi * 8;     // A-frag K sub-offset within 32-K step
  const int bkoff = hi * 8;     // B-frag K sub-offset within 32-K step

  v8f acc[4] = {v8f{}, v8f{}, v8f{}, v8f{}};
  const float* arow = X + (size_t)(mtile * 16 + l16) * HIDDEN;

  for (int k0 = 0; k0 < HIDDEN; k0 += 32) {
    // A fragment: row M = l16; elems 0..7 = K k0+abase.., 8..15 = K k0+16+abase..
    float at[16];
    *(float4*)(at + 0)  = *(const float4*)(arow + k0 + abase);
    *(float4*)(at + 4)  = *(const float4*)(arow + k0 + abase + 4);
    *(float4*)(at + 8)  = *(const float4*)(arow + k0 + 16 + abase);
    *(float4*)(at + 12) = *(const float4*)(arow + k0 + 16 + abase + 4);
    v16h a;
#pragma unroll
    for (int e = 0; e < 16; ++e) a[e] = (_Float16)at[e];

#pragma unroll
    for (int nt = 0; nt < 4; ++nt) {
      // B fragment: col N = l16 -> W row (ngrp*64 + nt*16 + l16), K k0+bkoff..+7, k0+16+bkoff..+7
      const float* brow =
          Wm + (size_t)(ngrp * 64 + nt * 16 + l16) * HIDDEN + k0 + bkoff;
      float bt[16];
#pragma unroll
      for (int q = 0; q < 4; ++q)
        *(float4*)(bt + 4 * q) = *(const float4*)(brow + 4 * q + (q >> 1) * 8);
      v16h bb;
#pragma unroll
      for (int e = 0; e < 16; ++e) bb[e] = (_Float16)bt[e];
      acc[nt] = wmma_f16(a, bb, acc[nt]);
    }
  }

  // D layout: VGPR r, lane -> (M = r + 8*hi, N = l16). Scatter to head layout
  // dst[bh][s][d], bh = b*NH + head, token m = s*BATCH + b.
#pragma unroll
  for (int nt = 0; nt < 4; ++nt) {
    const int col  = ngrp * 64 + nt * 16 + l16;
    const float bv_ = bias[col];
    const int d    = nt * 16 + l16;
#pragma unroll
    for (int r = 0; r < 8; ++r) {
      const int m = mtile * 16 + r + hi * 8;
      const int s = m >> 2;
      const int b = m & 3;
      dst[(((size_t)(b * NH + ngrp) * SEQ) + s) * HD + d] =
          (_Float16)(acc[nt][r] + bv_);
    }
  }
}

// ---------------------------------------------------------------------------
// Kernel 2: one wave per (head-batch bh, query tile of 16).
// ---------------------------------------------------------------------------
__global__ __launch_bounds__(32) void local_attn_wmma(
    const _Float16* __restrict__ qh, const _Float16* __restrict__ kh,
    const _Float16* __restrict__ vh, const unsigned char* __restrict__ mask,
    float* __restrict__ out, float* __restrict__ probs)
{
  // Vt: [HD][VSTR] f16 (V transposed, zero-padded cols 272..287)
  // Pl: [16][VSTR] f16 (probs in A-fragment-friendly layout, zero-padded)
  __shared__ _Float16 ldsbuf[(HD + 16) * VSTR];   // 47360 bytes
  _Float16* Vt = ldsbuf;
  _Float16* Pl = ldsbuf + HD * VSTR;

  const int lane = threadIdx.x & 31;
  const int l16  = lane & 15;
  const int hi   = lane >> 4;
  const int bh   = blockIdx.x >> 8;     // 64 head-batches
  const int qt   = blockIdx.x & 255;    // 256 query tiles
  const int s0   = qt * 16;
  const int b    = bh >> 4;
  const int h    = bh & 15;
  const int j0   = s0 - HALFW;          // key position of score-tile column 0

  const _Float16* vbase = vh + (size_t)bh * SEQ * HD;

  // ---- Zero Pl (v8h stores), stage V transposed into Vt -------------------
  {
    v8h z = {};
    for (int idx = lane; idx < (16 * VSTR) / 8; idx += 32)
      *((v8h*)Pl + idx) = z;

    // 1152 jobs: rowpair rp (0..143 -> j = 2*rp, 2*rp+1), d-chunk dc (0..7).
    // rp >= 136 (j >= 272) writes the K-padding zeros.
#pragma unroll 4
    for (int it = 0; it < 36; ++it) {
      const int idx = it * 32 + lane;
      const int rp  = idx >> 3;
      const int dc  = idx & 7;
      const int jj  = rp * 2;
      if (jj < 272) {
        int kp0 = j0 + jj, kp1 = j0 + jj + 1;
        kp0 = kp0 < 0 ? 0 : (kp0 > SEQ - 1 ? SEQ - 1 : kp0);
        kp1 = kp1 < 0 ? 0 : (kp1 > SEQ - 1 ? SEQ - 1 : kp1);
        const v8h r0 = *(const v8h*)(vbase + (size_t)kp0 * HD + dc * 8);
        const v8h r1 = *(const v8h*)(vbase + (size_t)kp1 * HD + dc * 8);
        __builtin_prefetch(vbase + (size_t)((kp1 + 64 < SEQ) ? kp1 + 64 : kp1) * HD, 0, 3);
#pragma unroll
        for (int e = 0; e < 8; ++e) {
          v2h pr; pr[0] = r0[e]; pr[1] = r1[e];
          *(v2h*)(Vt + (size_t)(dc * 8 + e) * VSTR + jj) = pr;
        }
      } else {
        const v2h zz = {};
#pragma unroll
        for (int e = 0; e < 8; ++e)
          *(v2h*)(Vt + (size_t)(dc * 8 + e) * VSTR + jj) = zz;
      }
    }
  }

  // ---- Q A-fragments (K = 64 dims -> two 16x16x32 steps) ------------------
  const _Float16* qrow = qh + ((size_t)bh * SEQ + (s0 + l16)) * HD;
  v16h a0, a1;
  {
    v8h t0 = *(const v8h*)(qrow + hi * 16);
    v8h t1 = *(const v8h*)(qrow + 8 + hi * 16);
    v8h t2 = *(const v8h*)(qrow + 32 + hi * 16);
    v8h t3 = *(const v8h*)(qrow + 40 + hi * 16);
#pragma unroll
    for (int e = 0; e < 8; ++e) {
      a0[e] = t0[e]; a0[8 + e] = t1[e];
      a1[e] = t2[e]; a1[8 + e] = t3[e];
    }
  }

  // ---- 16 queries x 272 keys score tile (17 WMMA column tiles) ------------
  float sc[17][8];
  bool  kval[17];
#pragma unroll
  for (int t = 0; t < 17; ++t) {
    const int kp  = j0 + t * 16 + l16;                      // this lane's key col
    const int kpc = kp < 0 ? 0 : (kp > SEQ - 1 ? SEQ - 1 : kp);
    const bool kv = (kp >= 0) && (kp < SEQ) &&
                    (mask[(size_t)kp * BATCH + b] == 0);
    const _Float16* krow = kh + ((size_t)bh * SEQ + kpc) * HD;
    v16h b0 = *(const v16h*)(krow + hi * 16);               // dims 0..31
    v16h b1 = *(const v16h*)(krow + 32 + hi * 16);          // dims 32..63
    v8f c = {};
    c = wmma_f16(a0, b0, c);
    c = wmma_f16(a1, b1, c);
    kval[t] = kv;
#pragma unroll
    for (int r = 0; r < 8; ++r) {
      const int i = r + hi * 8;                  // query row
      const int w = t * 16 + l16 - i;            // window offset index
      sc[t][r] = (kv && w >= 0 && w <= 256) ? c[r] : MIN_SCORE;
    }
  }

  // ---- Row softmax: per-register max/sum + half-wave shfl_xor reduction ---
  float mx[8], sm[8];
#pragma unroll
  for (int r = 0; r < 8; ++r) {
    float m = sc[0][r];
#pragma unroll
    for (int t = 1; t < 17; ++t) m = fmaxf(m, sc[t][r]);
    mx[r] = m;
  }
#pragma unroll
  for (int dmsk = 1; dmsk < 16; dmsk <<= 1)
#pragma unroll
    for (int r = 0; r < 8; ++r) mx[r] = fmaxf(mx[r], __shfl_xor(mx[r], dmsk, 32));

#pragma unroll
  for (int r = 0; r < 8; ++r) sm[r] = 0.0f;
#pragma unroll
  for (int t = 0; t < 17; ++t)
#pragma unroll
    for (int r = 0; r < 8; ++r) {
      const float e = __expf(sc[t][r] - mx[r]);  // MIN_SCORE -> exactly 0
      sc[t][r] = e;
      sm[r] += e;
    }
#pragma unroll
  for (int dmsk = 1; dmsk < 16; dmsk <<= 1)
#pragma unroll
    for (int r = 0; r < 8; ++r) sm[r] += __shfl_xor(sm[r], dmsk, 32);
#pragma unroll
  for (int r = 0; r < 8; ++r) sm[r] = 1.0f / sm[r];

  __syncthreads();   // Pl zero-fill complete before banded fill

  // ---- Probs: f32 band -> d_out, f16 -> Pl ---------------------------------
  float* prow = probs + (size_t)bh * SEQ * NWIN;
#pragma unroll
  for (int t = 0; t < 17; ++t) {
    const int j = t * 16 + l16;
#pragma unroll
    for (int r = 0; r < 8; ++r) {
      const int i = r + hi * 8;
      const int w = j - i;
      if (w >= 0 && w <= 256) {
        const float p = kval[t] ? sc[t][r] * sm[r] : 0.0f;
        prow[(size_t)(s0 + i) * NWIN + w] = p;
        Pl[(size_t)i * VSTR + j] = (_Float16)p;
      }
    }
  }
  __syncthreads();   // Pl band + Vt staging complete before fragment reads

  // ---- Aggregation via WMMA: out[16x64] = P[16x288] @ Vt^T[288x64] --------
  v8f agg[4] = {v8f{}, v8f{}, v8f{}, v8f{}};
#pragma unroll
  for (int ks = 0; ks < 9; ++ks) {
    const int kb = ks * 32;
    // A fragment of P: row i = l16, 16 contiguous K from kb + 16*hi (same as B)
    v16h aP;
    {
      const _Float16* pr_ = Pl + (size_t)l16 * VSTR + kb + hi * 16;
      v8h p0 = *(const v8h*)(pr_);
      v8h p1 = *(const v8h*)(pr_ + 8);
#pragma unroll
      for (int e = 0; e < 8; ++e) { aP[e] = p0[e]; aP[8 + e] = p1[e]; }
    }
#pragma unroll
    for (int nt = 0; nt < 4; ++nt) {
      // B fragment of V: col d = nt*16 + l16, 16 contiguous K from Vt row d
      const _Float16* vt = Vt + (size_t)(nt * 16 + l16) * VSTR + kb + hi * 16;
      v8h b0 = *(const v8h*)(vt);
      v8h b1 = *(const v8h*)(vt + 8);
      v16h bV;
#pragma unroll
      for (int e = 0; e < 8; ++e) { bV[e] = b0[e]; bV[8 + e] = b1[e]; }
      agg[nt] = wmma_f16(aP, bV, agg[nt]);
    }
  }

  // ---- Store out [SEQ, BATCH, HIDDEN]; D layout (M=r+8*hi, N=l16) ---------
#pragma unroll
  for (int nt = 0; nt < 4; ++nt) {
    const int d = nt * 16 + l16;
#pragma unroll
    for (int r = 0; r < 8; ++r) {
      const int i = r + hi * 8;
      out[((size_t)(s0 + i) * BATCH + b) * HIDDEN + h * HD + d] = agg[nt][r];
    }
  }
}

// ---------------------------------------------------------------------------
extern "C" void kernel_launch(void* const* d_in, const int* in_sizes, int n_in,
                              void* d_out, int out_size, void* d_ws,
                              size_t ws_size, hipStream_t stream) {
  const float* q_in = (const float*)d_in[0];
  const float* k_in = (const float*)d_in[1];
  const float* v_in = (const float*)d_in[2];
  const unsigned char* mask = (const unsigned char*)d_in[3];
  const float* Wq = (const float*)d_in[4];
  const float* bq = (const float*)d_in[5];
  const float* Wk = (const float*)d_in[6];
  const float* bk = (const float*)d_in[7];
  const float* Wv = (const float*)d_in[8];
  const float* bv = (const float*)d_in[9];

  const size_t headElems = (size_t)NBH * SEQ * HD;   // 16,777,216 f16 each
  _Float16* qh = (_Float16*)d_ws;
  _Float16* kh = qh + headElems;
  _Float16* vh = kh + headElems;

  float* out   = (float*)d_out;                       // [SEQ, BATCH, HIDDEN]
  float* probs = out + (size_t)SEQ * BATCH * HIDDEN;  // [NBH, SEQ, NWIN]

  // 3 matrices * 1024 mtiles * 16 head-groups = 49152 waves, 8 waves/block
  qkv_proj_wmma<<<6144, 256, 0, stream>>>(q_in, k_in, v_in, Wq, bq, Wk, bk,
                                          Wv, bv, qh, kh, vh);
  // 64 head-batches * 256 query tiles, one wave each
  local_attn_wmma<<<NBH * 256, 32, 0, stream>>>(qh, kh, vh, mask, out, probs);
}
